// WindowAttention_87376814670114
// MI455X (gfx1250) — matrix operand order repaired and checked by reading, not compiled, not run on hardware
//
#include <hip/hip_runtime.h>

// ---------------------------------------------------------------------------
// SwinV2 window attention, fused, bf16 WMMA (gfx1250 / CDNA5, wave32)
// B=4096 windows, N=64 tokens, C=256, H=8 heads, hd=32
// ---------------------------------------------------------------------------

typedef __attribute__((ext_vector_type(16))) __bf16 v16bf;
typedef __attribute__((ext_vector_type(2)))  __bf16 v2bf;
typedef __attribute__((ext_vector_type(8)))  float  v8f;

union V16 { unsigned int u[8]; v16bf v; };
union V8  { float f[8]; v8f v; };

__device__ __forceinline__ unsigned short f2bf(float x) {
  unsigned int u = __float_as_uint(x);
  u += 0x7FFFu + ((u >> 16) & 1u);          // round-to-nearest-even
  return (unsigned short)(u >> 16);
}

// Pack two f32 -> bf16x2 in one u32 (hardware pk convert when available).
__device__ __forceinline__ unsigned int f2bf2(float a, float b) {
#if defined(__has_builtin) && __has_builtin(__builtin_amdgcn_cvt_pk_bf16_f32)
  union { v2bf v; unsigned int u; } t;
  t.v = __builtin_amdgcn_cvt_pk_bf16_f32(a, b);
  return t.u;
#else
  return (unsigned int)f2bf(a) | ((unsigned int)f2bf(b) << 16);
#endif
}

// A-matrix 16x32 bf16 fragment (M x K), src row-major [rows][stride] (elems).
__device__ __forceinline__ v16bf load_a(const unsigned short* src, int stride,
                                        int mbase, int kbase, int r, int hi) {
  V16 t;
  int base = (mbase + r) * stride + kbase;
#pragma unroll
  for (int i = 0; i < 4; ++i) {
    t.u[i]     = *(const unsigned int*)(src + base + hi * 8 + 2 * i);
    t.u[i + 4] = *(const unsigned int*)(src + base + 16 + hi * 8 + 2 * i);
  }
  return t.v;
}

// B-matrix 32x16 bf16 fragment (K x N), source stored as src[n][k] row-major.
__device__ __forceinline__ v16bf load_b(const unsigned short* src, int stride,
                                        int nbase, int kbase, int r, int hi) {
  V16 t;
  int base = (nbase + r) * stride + kbase;
#pragma unroll
  for (int j = 0; j < 4; ++j) {
    t.u[j]     = *(const unsigned int*)(src + base + hi * 8 + 2 * j);
    t.u[j + 4] = *(const unsigned int*)(src + base + 16 + hi * 8 + 2 * j);
  }
  return t.v;
}

__device__ __forceinline__ v8f wmma_bf16(v16bf a, v16bf b, v8f c) {
  return __builtin_amdgcn_wmma_f32_16x16x32_bf16(false, a, false, b,
                                                 (short)0, c, false, false);
}

// ---------------------------------------------------------------------------
// Kernel 0: convert f32 weights -> bf16 in workspace
// ---------------------------------------------------------------------------
__global__ void wa_convert_weights(const float* __restrict__ qkv_w,
                                   const float* __restrict__ proj_w,
                                   unsigned int* __restrict__ wqkv,
                                   unsigned int* __restrict__ wproj) {
  int i = blockIdx.x * 256 + threadIdx.x;        // u32-pair index
  if (i < 98304) wqkv[i] = f2bf2(qkv_w[2 * i], qkv_w[2 * i + 1]);
  if (i < 32768) wproj[i] = f2bf2(proj_w[2 * i], proj_w[2 * i + 1]);
}

// ---------------------------------------------------------------------------
// Kernel 1: SwinV2 continuous position bias -> f32 [8][64][64] in workspace
// ---------------------------------------------------------------------------
__global__ void wa_bias(const float* __restrict__ w1,   // [512][2]
                        const float* __restrict__ b1,   // [512]
                        const float* __restrict__ w2,   // [8][512]
                        float* __restrict__ bias) {     // [8][64][64]
  const int i = blockIdx.x;    // 64
  const int j = threadIdx.x;   // 64
  const float dh = (float)((i >> 3) - (j >> 3));
  const float dw = (float)((i & 7) - (j & 7));
  const float sh = dh * (8.0f / 7.0f);
  const float sw = dw * (8.0f / 7.0f);
  const float vh = ((sh > 0.f) - (sh < 0.f)) * __log2f(fabsf(sh) + 1.0f) * (1.0f / 3.0f);
  const float vw = ((sw > 0.f) - (sw < 0.f)) * __log2f(fabsf(sw) + 1.0f) * (1.0f / 3.0f);

  float acc[8];
#pragma unroll
  for (int h = 0; h < 8; ++h) acc[h] = 0.0f;
  for (int k = 0; k < 512; ++k) {
    float hv = vh * w1[2 * k] + vw * w1[2 * k + 1] + b1[k];
    hv = fmaxf(hv, 0.0f);
#pragma unroll
    for (int h = 0; h < 8; ++h) acc[h] += hv * w2[h * 512 + k];
  }
#pragma unroll
  for (int h = 0; h < 8; ++h)
    bias[h * 4096 + i * 64 + j] = 16.0f / (1.0f + __expf(-acc[h]));
}

// ---------------------------------------------------------------------------
// Kernel 2: fused window attention. 1 block = 1 window, 8 waves, 1 head/wave.
// LDS layout (bf16 elems):
//   [0 .. 16896)        xs : x (64 x 256, stride 264); reused later for O
//   [16896 .. 78336)    per head h (7680 elems):
//       q : 64 x 32 (stride 40)   at +0
//       k : 64 x 32 (stride 40)   at +2560
//       v : 32 x 64 transposed (stride 72) at +5120
//       P : 64 x 64 (stride 72) overlays q+k after scores are consumed
// ---------------------------------------------------------------------------
#define XS_STR 264
#define QKV_OFF 16896
#define LDS_ELEMS 78336

__global__ void __launch_bounds__(256)
wa_attn(const float* __restrict__ x,
        const unsigned short* __restrict__ wqkv,   // bf16 [768][256]
        const unsigned short* __restrict__ wproj,  // bf16 [256][256]
        const float* __restrict__ bias,            // [8][64][64]
        const float* __restrict__ proj_b,          // [256]
        float* __restrict__ out) {                 // [B][64][256]
  __shared__ __align__(16) unsigned short lds[LDS_ELEMS];

  const int b    = blockIdx.x;
  const int tid  = threadIdx.x;
  const int wave = tid >> 5;
  const int lane = tid & 31;
  const int r    = lane & 15;
  const int hi   = lane >> 4;

  // ---- stage 1: load x -> bf16 LDS --------------------------------------
  const float* xb = x + (size_t)b * 16384;
#pragma unroll
  for (int it = 0; it < 16; ++it) {
    int i  = tid + it * 256;           // float4 index in [0,4096)
    int n  = i >> 6;                   // 64 float4 per row
    int c4 = (i & 63) * 4;
    float4 v = ((const float4*)xb)[i];
    unsigned int* dst = (unsigned int*)(lds + n * XS_STR + c4);
    dst[0] = f2bf2(v.x, v.y);
    dst[1] = f2bf2(v.z, v.w);
  }
  __syncthreads();

  // ---- stage 2a: q,k = x @ Wqkv^T[:, 0:512]  (branchless store) ----------
  for (int t = wave; t < 128; t += 8) {
    int mi = t & 3, ni = t >> 2;       // ni in [0,32)
    V8 acc;
#pragma unroll
    for (int i = 0; i < 8; ++i) acc.f[i] = 0.0f;
#pragma unroll
    for (int kk = 0; kk < 8; ++kk) {
      v16bf a  = load_a(lds, XS_STR, mi * 16, kk * 32, r, hi);
      v16bf bw = load_b(wqkv, 256, ni * 16, kk * 32, r, hi);
      acc.v = wmma_bf16(a, bw, acc.v);
    }
    int obase = ni * 16;               // [0,512)
    int s   = obase >> 8;              // 0=q, 1=k
    int h   = (obase >> 5) & 7;
    int col = (obase & 31) + r;
    int base = QKV_OFF + h * 7680 + s * 2560 + col;
#pragma unroll
    for (int i = 0; i < 8; ++i) {
      int row = mi * 16 + i + hi * 8;
      lds[base + row * 40] = f2bf(acc.f[i]);
    }
  }
  // ---- stage 2b: v (stored transposed, pair-packed stores) ---------------
  for (int t = wave; t < 64; t += 8) {
    int mi = t & 3, niv = t >> 2;      // niv in [0,16)
    V8 acc;
#pragma unroll
    for (int i = 0; i < 8; ++i) acc.f[i] = 0.0f;
#pragma unroll
    for (int kk = 0; kk < 8; ++kk) {
      v16bf a  = load_a(lds, XS_STR, mi * 16, kk * 32, r, hi);
      v16bf bw = load_b(wqkv, 256, 512 + niv * 16, kk * 32, r, hi);
      acc.v = wmma_bf16(a, bw, acc.v);
    }
    int h   = niv >> 1;
    int col = (niv & 1) * 16 + r;      // channel d in [0,32)
    int vbase = QKV_OFF + h * 7680 + 5120 + col * 72;
#pragma unroll
    for (int ii = 0; ii < 4; ++ii) {
      int row0 = mi * 16 + 2 * ii + hi * 8;   // even token row
      *(unsigned int*)(lds + vbase + row0) = f2bf2(acc.f[2 * ii], acc.f[2 * ii + 1]);
    }
  }
  __syncthreads();

  // ---- stage 3: per-head scores + softmax (head = wave) ------------------
  {
    const int h  = wave;
    const int qb = QKV_OFF + h * 7680;
    const int kb = qb + 2560;
    V8 sacc[4][4];
    v16bf aq[4];
#pragma unroll
    for (int mi = 0; mi < 4; ++mi)
      aq[mi] = load_a(lds + qb, 40, mi * 16, 0, r, hi);
#pragma unroll
    for (int ni = 0; ni < 4; ++ni) {
      v16bf bk = load_b(lds + kb, 40, ni * 16, 0, r, hi);
#pragma unroll
      for (int mi = 0; mi < 4; ++mi) {
        V8 z;
#pragma unroll
        for (int i = 0; i < 8; ++i) z.f[i] = 0.0f;
        sacc[mi][ni].v = wmma_bf16(aq[mi], bk, z.v);
      }
    }
    const float scale = 0.17677669529663687f;  // 1/sqrt(32)
    const float* bh = bias + h * 4096;
#pragma unroll
    for (int mi = 0; mi < 4; ++mi) {
#pragma unroll
      for (int i = 0; i < 8; ++i) {
        int row = mi * 16 + i + hi * 8;
        float e[4];
        float mx = -3.4e38f;
#pragma unroll
        for (int ni = 0; ni < 4; ++ni) {
          float sv = sacc[mi][ni].f[i] * scale + bh[row * 64 + ni * 16 + r];
          e[ni] = sv;
          mx = fmaxf(mx, sv);
        }
        mx = fmaxf(mx, __shfl_xor(mx, 1, 32));
        mx = fmaxf(mx, __shfl_xor(mx, 2, 32));
        mx = fmaxf(mx, __shfl_xor(mx, 4, 32));
        mx = fmaxf(mx, __shfl_xor(mx, 8, 32));
        float sum = 0.0f;
#pragma unroll
        for (int ni = 0; ni < 4; ++ni) { e[ni] = __expf(e[ni] - mx); sum += e[ni]; }
        sum += __shfl_xor(sum, 1, 32);
        sum += __shfl_xor(sum, 2, 32);
        sum += __shfl_xor(sum, 4, 32);
        sum += __shfl_xor(sum, 8, 32);
        float inv = 1.0f / sum;
#pragma unroll
        for (int ni = 0; ni < 4; ++ni)
          lds[qb + row * 72 + ni * 16 + r] = f2bf(e[ni] * inv);  // P overlays q/k
      }
    }
  }
  __syncthreads();

  // ---- stage 4: O = P @ V  (64x64 @ 64x32), O staged into xs region ------
  {
    const int h  = wave;
    const int pb = QKV_OFF + h * 7680;
    const int vb = pb + 5120;
    V8 oacc[4][2];
#pragma unroll
    for (int mi = 0; mi < 4; ++mi)
#pragma unroll
      for (int ni = 0; ni < 2; ++ni)
#pragma unroll
        for (int i = 0; i < 8; ++i) oacc[mi][ni].f[i] = 0.0f;
#pragma unroll
    for (int kk = 0; kk < 2; ++kk) {
      v16bf ap[4];
#pragma unroll
      for (int mi = 0; mi < 4; ++mi)
        ap[mi] = load_a(lds + pb, 72, mi * 16, kk * 32, r, hi);
#pragma unroll
      for (int ni = 0; ni < 2; ++ni) {
        v16bf bv = load_b(lds + vb, 72, ni * 16, kk * 32, r, hi);  // v^T[d][m]
#pragma unroll
        for (int mi = 0; mi < 4; ++mi)
          oacc[mi][ni].v = wmma_bf16(ap[mi], bv, oacc[mi][ni].v);
      }
    }
#pragma unroll
    for (int mi = 0; mi < 4; ++mi)
#pragma unroll
      for (int ni = 0; ni < 2; ++ni)
#pragma unroll
        for (int i = 0; i < 8; ++i) {
          int row = mi * 16 + i + hi * 8;
          int ch  = h * 32 + ni * 16 + r;
          lds[row * XS_STR + ch] = f2bf(oacc[mi][ni].f[i]);
        }
  }
  __syncthreads();

  // ---- stage 5: out = O @ Wproj^T + b  (64x256 @ 256x256) ----------------
  float* ob = out + (size_t)b * 16384;
  for (int t = wave; t < 64; t += 8) {
    int mi = t & 3, ni = t >> 2;
    V8 acc;
#pragma unroll
    for (int i = 0; i < 8; ++i) acc.f[i] = 0.0f;
#pragma unroll
    for (int kk = 0; kk < 8; ++kk) {
      v16bf a  = load_a(lds, XS_STR, mi * 16, kk * 32, r, hi);
      v16bf bw = load_b(wproj, 256, ni * 16, kk * 32, r, hi);
      acc.v = wmma_bf16(a, bw, acc.v);
    }
    int col = ni * 16 + r;
    float pbias = proj_b[col];
#pragma unroll
    for (int i = 0; i < 8; ++i) {
      int row = mi * 16 + i + hi * 8;
      ob[row * 256 + col] = acc.f[i] + pbias;
    }
  }
}

// ---------------------------------------------------------------------------
extern "C" void kernel_launch(void* const* d_in, const int* in_sizes, int n_in,
                              void* d_out, int out_size, void* d_ws, size_t ws_size,
                              hipStream_t stream) {
  const float* x      = (const float*)d_in[0];  // [4096,64,256]
  const float* qkv_w  = (const float*)d_in[1];  // [768,256]
  const float* proj_w = (const float*)d_in[2];  // [256,256]
  const float* proj_b = (const float*)d_in[3];  // [256]
  const float* cpb_w1 = (const float*)d_in[4];  // [512,2]
  const float* cpb_b1 = (const float*)d_in[5];  // [512]
  const float* cpb_w2 = (const float*)d_in[6];  // [8,512]
  float* out = (float*)d_out;

  char* ws = (char*)d_ws;
  float*          bias  = (float*)ws;                       // 131072 B
  unsigned short* wqkv  = (unsigned short*)(ws + 131072);   // 393216 B
  unsigned short* wproj = (unsigned short*)(ws + 524288);   // 131072 B

  wa_convert_weights<<<1024, 256, 0, stream>>>(qkv_w, proj_w,
                                               (unsigned int*)wqkv,
                                               (unsigned int*)wproj);
  wa_bias<<<64, 64, 0, stream>>>(cpb_w1, cpb_b1, cpb_w2, bias);
  wa_attn<<<4096, 256, 0, stream>>>(x, wqkv, wproj, bias, proj_b, out);
}
